// RelPositionMultiHeadAttentionLongformer_39659728011631
// MI455X (gfx1250) — matrix operand order repaired and checked by reading, not compiled, not run on hardware
//
#include <hip/hip_runtime.h>

// ---------------------------------------------------------------------------
// Longformer rel-pos attention for MI455X (gfx1250).
// B=4 T=2048 NF=512 H=8 DK=64 W=128 band=257.
// All matrix math on v_wmma_f32_16x16x32_bf16 (f32 accumulate).
// All intermediates stored bf16 (halves HBM/L2 traffic); LDS tiles staged with
// GLOBAL_LOAD_ASYNC_TO_LDS_B128 (ASYNCcnt) when the builtin exists.
// ---------------------------------------------------------------------------

constexpr int BB    = 4;
constexpr int TT    = 2048;
constexpr int NHEAD = 8;
constexpr int DKC   = 64;
constexpr int NFC   = 512;
constexpr int WC    = 128;
constexpr int BAND  = 257;   // 2*W+1
constexpr int QBC   = 32;    // query rows per attention workgroup
constexpr int NKC   = 320;   // padded key columns per block (band span 288 -> 320)
constexpr int NKP   = 328;   // score row stride in floats (padding)

typedef __bf16 v16bf __attribute__((ext_vector_type(16)));
typedef float  v8f   __attribute__((ext_vector_type(8)));
typedef float  f32x4 __attribute__((ext_vector_type(4)));
typedef unsigned int u32x4 __attribute__((ext_vector_type(4)));
typedef int    i32x4 __attribute__((ext_vector_type(4)));

typedef __attribute__((address_space(1))) i32x4 as1_i32x4;
typedef __attribute__((address_space(3))) i32x4 as3_i32x4;

union Frag { v16bf v; u32x4 u[2]; };

#if __has_builtin(__builtin_amdgcn_global_load_async_to_lds_b128)
#define HAS_ASYNC_LDS 1
#else
#define HAS_ASYNC_LDS 0
#endif

// Copy 16 bytes (8 bf16) global -> LDS. Async (ASYNCcnt) when available.
__device__ __forceinline__ void copy16_g2l(const __bf16* g, __bf16* l) {
#if HAS_ASYNC_LDS
  __builtin_amdgcn_global_load_async_to_lds_b128(
      (as1_i32x4*)(const_cast<__bf16*>(g)),
      (as3_i32x4*)(l), 0, 0);
#else
  *(u32x4*)l = *(const u32x4*)g;
#endif
}

__device__ __forceinline__ void copies_join() {
#if HAS_ASYNC_LDS && __has_builtin(__builtin_amdgcn_s_wait_asynccnt)
  __builtin_amdgcn_s_wait_asynccnt(0);
#endif
}

__device__ __forceinline__ void zero16_l(__bf16* l) {
  u32x4 z = {0u, 0u, 0u, 0u};
  *(u32x4*)l = z;
}

__device__ __forceinline__ v8f vzero8() {
  v8f v = {0.f, 0.f, 0.f, 0.f, 0.f, 0.f, 0.f, 0.f};
  return v;
}

__device__ __forceinline__ v8f wmma_bf16(v16bf a, v16bf b, v8f c) {
  // D = A(16x32 bf16) * B(32x16 bf16) + C(16x16 f32)
  return __builtin_amdgcn_wmma_f32_16x16x32_bf16(false, a, false, b, (short)0, c,
                                                 false, false);
}

// A-style fragment from bf16 memory: elements 0..7 = p[0..7], 8..15 = p[16..23]
__device__ __forceinline__ v16bf frag_from_bf16(const __bf16* p) {
  Frag f;
  f.u[0] = *(const u32x4*)(p);
  f.u[1] = *(const u32x4*)(p + 16);
  return f.v;
}

// A-style fragment from f32 memory (used on the LDS score tile).
__device__ __forceinline__ v16bf frag_from_f32(const float* p) {
  f32x4 a = ((const f32x4*)p)[0];
  f32x4 b = ((const f32x4*)p)[1];
  f32x4 c = ((const f32x4*)(p + 16))[0];
  f32x4 d = ((const f32x4*)(p + 16))[1];
  v16bf v;
#pragma unroll
  for (int i = 0; i < 4; ++i) {
    v[i]      = (__bf16)a[i];
    v[4 + i]  = (__bf16)b[i];
    v[8 + i]  = (__bf16)c[i];
    v[12 + i] = (__bf16)d[i];
  }
  return v;
}

// ---------------------------------------------------------------------------
// f32 -> bf16 bulk convert (inputs & weights, done once).
// ---------------------------------------------------------------------------
__global__ void cvt_f32_to_bf16(const float* __restrict__ src,
                                __bf16* __restrict__ dst, int n) {
  const int i = (blockIdx.x * blockDim.x + threadIdx.x) * 8;
  if (i < n) {   // n is a multiple of 8
    f32x4 a = *(const f32x4*)(src + i);
    f32x4 b = *(const f32x4*)(src + i + 4);
    union { __bf16 h[8]; u32x4 u; } t;
#pragma unroll
    for (int j = 0; j < 4; ++j) { t.h[j] = (__bf16)a[j]; t.h[4 + j] = (__bf16)b[j]; }
    *(u32x4*)(dst + i) = t.u;
  }
}

// Fold bq + pos_bias_u / pos_bias_v into per-feature bias vectors.
__global__ void bias_combine(const float* __restrict__ bq,
                             const float* __restrict__ u,
                             const float* __restrict__ v,
                             float* __restrict__ bu, float* __restrict__ bv) {
  const int i = threadIdx.x;   // 512 threads
  bu[i] = bq[i] + u[i];
  bv[i] = bq[i] + v[i];
}

// ---------------------------------------------------------------------------
// C(MxN) = A(MxK) @ W(NxK)^T + bias.  A,W bf16; C f32 or bf16 (template).
// Block 128x128, 256 threads = 8 waves (wave32), wave tile 32x64, BK=32.
// Double-buffered LDS, async global->LDS staging.
// ---------------------------------------------------------------------------
template <typename TC>
__global__ __launch_bounds__(256) void gemm_bf16(
    const __bf16* __restrict__ A, int lda,
    const __bf16* __restrict__ Wt, int ldw,
    const float* __restrict__ bias,
    TC* __restrict__ C, int ldc,
    int M, int N, int K)
{
  __shared__ __bf16 As[2][128 * 32];
  __shared__ __bf16 Bs[2][128 * 32];

  const int tid  = threadIdx.x;
  const int wid  = tid >> 5;
  const int lane = tid & 31;
  const int ln   = lane & 15;
  const int mhi  = (lane < 16) ? 0 : 8;
  const int kbA  = (lane < 16) ? 0 : 8;   // A-fragment K base
  const int kbB  = (lane < 16) ? 0 : 16;  // unused; B fragment also uses kbA
  const int bm = blockIdx.y * 128, bn = blockIdx.x * 128;
  const int wm = (wid & 3) * 32, wn = (wid >> 2) * 64;

  const int srow = tid & 127;
  const int sh16 = (tid >> 7) * 16;

  auto stage = [&](int buf, int k0) {
    const __bf16* ga = A + (size_t)(bm + srow) * lda + k0 + sh16;
    __bf16* la = &As[buf][srow * 32 + sh16];
    if ((bm + srow) < M) { copy16_g2l(ga, la); copy16_g2l(ga + 8, la + 8); }
    else                 { zero16_l(la);       zero16_l(la + 8); }
    const __bf16* gb = Wt + (size_t)(bn + srow) * ldw + k0 + sh16;
    __bf16* lb = &Bs[buf][srow * 32 + sh16];
    if ((bn + srow) < N) { copy16_g2l(gb, lb); copy16_g2l(gb + 8, lb + 8); }
    else                 { zero16_l(lb);       zero16_l(lb + 8); }
  };

  v8f acc[2][4];
#pragma unroll
  for (int mt = 0; mt < 2; ++mt)
#pragma unroll
    for (int nt = 0; nt < 4; ++nt) acc[mt][nt] = vzero8();

  stage(0, 0);
  int buf = 0;
  for (int k0 = 0; k0 < K; k0 += 32) {
    copies_join();
    __syncthreads();                 // tile k0 ready; prior compute done
    if (k0 + 32 < K) stage(buf ^ 1, k0 + 32);

    Frag af[2], bf[4];
#pragma unroll
    for (int mt = 0; mt < 2; ++mt) {
      const __bf16* p = &As[buf][(wm + mt * 16 + ln) * 32 + kbA];
      af[mt].u[0] = *(const u32x4*)(p);
      af[mt].u[1] = *(const u32x4*)(p + 16);
    }
#pragma unroll
    for (int nt = 0; nt < 4; ++nt) {
      const __bf16* p = &Bs[buf][(wn + nt * 16 + ln) * 32 + kbA];
      bf[nt].u[0] = *(const u32x4*)(p);
      bf[nt].u[1] = *(const u32x4*)(p + 16);
    }
#pragma unroll
    for (int mt = 0; mt < 2; ++mt)
#pragma unroll
      for (int nt = 0; nt < 4; ++nt)
        acc[mt][nt] = wmma_bf16(af[mt].v, bf[nt].v, acc[mt][nt]);
    buf ^= 1;
  }

#pragma unroll
  for (int mt = 0; mt < 2; ++mt)
#pragma unroll
    for (int nt = 0; nt < 4; ++nt) {
      const int col = bn + wn + nt * 16 + ln;
      float bv = 0.f;
      if (bias != nullptr && col < N) bv = bias[col];
#pragma unroll
      for (int r = 0; r < 8; ++r) {
        const int m = bm + wm + mt * 16 + mhi + r;
        if (m < M && col < N) C[(size_t)m * ldc + col] = (TC)(acc[mt][nt][r] + bv);
      }
    }
}

__device__ __forceinline__ void load_q_frags(Frag aq[2][2], const __bf16* Q,
                                             int bI, int h, int q0, int ln, int kbA) {
#pragma unroll
  for (int mt = 0; mt < 2; ++mt)
#pragma unroll
    for (int ks = 0; ks < 2; ++ks)
      aq[mt][ks].v = frag_from_bf16(
          Q + (size_t)(bI * TT + q0 + mt * 16 + ln) * NFC + h * DKC + ks * 32 + kbA);
}

// ---------------------------------------------------------------------------
// Banded attention: one workgroup (128 threads, 4 waves) per 32 query rows per
// (b, h).  Keys j = q0-128+kk, kk in [0,320); band offset d = kk - qi (0..256).
// S and BD via WMMA into an LDS f32 score tile; softmax in LDS; P@V via WMMA.
// ---------------------------------------------------------------------------
__global__ __launch_bounds__(128) void attn_kernel(
    const __bf16* __restrict__ QU, const __bf16* __restrict__ QV,
    const __bf16* __restrict__ KW, const __bf16* __restrict__ VW,
    const __bf16* __restrict__ PW, const unsigned char* __restrict__ pad,
    __bf16* __restrict__ CTX)
{
  extern __shared__ char smem[];
  float*  sc  = (float*)smem;                                        // [32][NKP]
  __bf16* sh  = (__bf16*)(smem + (size_t)QBC * NKP * 4);             // 320*64 bf16 (reused 3x)
  float*  red = (float*)(smem + (size_t)QBC * NKP * 4 + (size_t)NKC * DKC * 2);  // [256]

  const int tid  = threadIdx.x;
  const int wid  = tid >> 5;
  const int lane = tid & 31;
  const int ln   = lane & 15;
  const int mhi  = (lane < 16) ? 0 : 8;
  const int kbA  = (lane < 16) ? 0 : 8;
  const int kbB  = (lane < 16) ? 0 : 16;
  const int bI = blockIdx.z, h = blockIdx.y;
  const int q0 = blockIdx.x * QBC;

  // ---- Phase 0: stage K tile  sh[kk*64 + dd]  (async; zeros out of range) ----
  for (int idx = tid; idx < NKC * 8; idx += 128) {
    const int kk = idx >> 3, sub = (idx & 7) * 8;
    const int j = q0 - WC + kk;
    __bf16* l = sh + kk * DKC + sub;
    if (j >= 0 && j < TT)
      copy16_g2l(KW + (size_t)(bI * TT + j) * NFC + h * DKC + sub, l);
    else
      zero16_l(l);
  }
  copies_join();
  __syncthreads();

  // ---- Phase 1: S = QU_block(32x64) @ K3^T -> sc[32][320] ----
  {
    Frag aq[2][2];
    load_q_frags(aq, QU, bI, h, q0, ln, kbA);
    v8f acc[2][5];
#pragma unroll
    for (int mt = 0; mt < 2; ++mt)
#pragma unroll
      for (int ci = 0; ci < 5; ++ci) acc[mt][ci] = vzero8();

#pragma unroll
    for (int ks = 0; ks < 2; ++ks)
#pragma unroll
      for (int ci = 0; ci < 5; ++ci) {
        const int kk = (wid * 5 + ci) * 16 + ln;
        Frag bf;
        const __bf16* p = sh + kk * DKC + ks * 32 + kbA;
        bf.u[0] = *(const u32x4*)(p);
        bf.u[1] = *(const u32x4*)(p + 16);
#pragma unroll
        for (int mt = 0; mt < 2; ++mt)
          acc[mt][ci] = wmma_bf16(aq[mt][ks].v, bf.v, acc[mt][ci]);
      }
#pragma unroll
    for (int mt = 0; mt < 2; ++mt)
#pragma unroll
      for (int ci = 0; ci < 5; ++ci) {
        const int col = (wid * 5 + ci) * 16 + ln;
#pragma unroll
        for (int r = 0; r < 8; ++r)
          sc[(mt * 16 + mhi + r) * NKP + col] = acc[mt][ci][r];
      }
  }
  __syncthreads();

  // ---- Phase 2: stage pos tile  sh[d*64 + dd]  (async; zeros d >= 257) ----
  for (int idx = tid; idx < 272 * 8; idx += 128) {
    const int d = idx >> 3, sub = (idx & 7) * 8;
    __bf16* l = sh + d * DKC + sub;
    if (d < BAND)
      copy16_g2l(PW + (size_t)d * NFC + h * DKC + sub, l);
    else
      zero16_l(l);
  }
  copies_join();
  __syncthreads();

  // ---- Phase 3: BD = QV_block @ P^T, scatter-add at kk = d + qi ----
  {
    const int st = (wid == 3) ? 14 : wid * 5;       // d-tile starts 0,5,10,14
    const int cn = 5 - ((wid > 1) ? (wid - 1) : 0); // counts 5,5,4,3 (17 tiles)
    Frag aq[2][2];
    load_q_frags(aq, QV, bI, h, q0, ln, kbA);
    v8f acc[2][5];
#pragma unroll
    for (int mt = 0; mt < 2; ++mt)
#pragma unroll
      for (int ci = 0; ci < 5; ++ci) acc[mt][ci] = vzero8();

#pragma unroll
    for (int ks = 0; ks < 2; ++ks)
#pragma unroll
      for (int ci = 0; ci < 5; ++ci) {
        if (ci < cn) {
          const int dt = (st + ci) * 16 + ln;
          Frag bf;
          const __bf16* p = sh + dt * DKC + ks * 32 + kbA;
          bf.u[0] = *(const u32x4*)(p);
          bf.u[1] = *(const u32x4*)(p + 16);
#pragma unroll
          for (int mt = 0; mt < 2; ++mt)
            acc[mt][ci] = wmma_bf16(aq[mt][ks].v, bf.v, acc[mt][ci]);
        }
      }
#pragma unroll
    for (int mt = 0; mt < 2; ++mt)
#pragma unroll
      for (int ci = 0; ci < 5; ++ci) {
        if (ci < cn) {
          const int d = (st + ci) * 16 + ln;
#pragma unroll
          for (int r = 0; r < 8; ++r) {
            const int m = mt * 16 + mhi + r;
            sc[m * NKP + d + m] += acc[mt][ci][r];  // unique (m,kk) per element
          }
        }
      }
  }
  __syncthreads();

  // ---- Phase 4: mask + softmax over kk (4 threads per row, 80 cols each) ----
  {
    const int row = tid >> 2, part = tid & 3;
    const int tg = q0 + row;
    const bool padq = pad[(size_t)bI * TT + tg] != 0;
    const int c0 = part * 80;
    float* srow = sc + row * NKP;

    float mx = -1e30f;
    for (int c = c0; c < c0 + 80; ++c) {
      const int d = c - row;
      const int j = q0 - WC + c;
      bool valid = (d >= 0) && (d <= 2 * WC) && (j >= 0) && (j < TT);
      if (valid && pad[(size_t)bI * TT + j]) valid = false;
      if (valid) mx = fmaxf(mx, srow[c] * 0.125f);
    }
    red[row * 4 + part] = mx;
    __syncthreads();
    const float rm = fmaxf(fmaxf(red[row * 4 + 0], red[row * 4 + 1]),
                           fmaxf(red[row * 4 + 2], red[row * 4 + 3]));
    float sum = 0.f;
    for (int c = c0; c < c0 + 80; ++c) {
      const int d = c - row;
      const int j = q0 - WC + c;
      bool valid = (d >= 0) && (d <= 2 * WC) && (j >= 0) && (j < TT);
      if (valid && pad[(size_t)bI * TT + j]) valid = false;
      float e = 0.f;
      if (valid) e = __expf(srow[c] * 0.125f - rm);
      srow[c] = e;
      sum += e;
    }
    red[128 + row * 4 + part] = sum;
    __syncthreads();
    const float tot = red[128 + row * 4 + 0] + red[128 + row * 4 + 1] +
                      red[128 + row * 4 + 2] + red[128 + row * 4 + 3];
    float inv = (tot > 0.f) ? (1.f / tot) : 0.f;
    if (padq) inv = 0.f;
    for (int c = c0; c < c0 + 80; ++c) srow[c] *= inv;
  }
  __syncthreads();

  // ---- Phase 5: stage V transposed  sh[dd*NKC + kk] (sync scatter) ----
  for (int idx = tid; idx < NKC * 8; idx += 128) {
    const int kk = idx >> 3, seg = (idx & 7) * 8;
    const int j = q0 - WC + kk;
    union { u32x4 u; __bf16 hv[8]; } t;
    u32x4 z = {0u, 0u, 0u, 0u};
    t.u = z;
    if (j >= 0 && j < TT)
      t.u = *(const u32x4*)(VW + (size_t)(bI * TT + j) * NFC + h * DKC + seg);
#pragma unroll
    for (int i = 0; i < 8; ++i) sh[(seg + i) * NKC + kk] = t.hv[i];
  }
  __syncthreads();

  // ---- Phase 6: O(32x64) = P(32x320) @ V3(320x64); wave -> one dd tile ----
  {
    v8f acc[2];
    acc[0] = vzero8();
    acc[1] = vzero8();
#pragma unroll
    for (int ks = 0; ks < 10; ++ks) {
      Frag bf;
      const __bf16* pB = sh + (wid * 16 + ln) * NKC + ks * 32 + kbA;
      bf.u[0] = *(const u32x4*)(pB);
      bf.u[1] = *(const u32x4*)(pB + 16);
#pragma unroll
      for (int mt = 0; mt < 2; ++mt) {
        v16bf av = frag_from_f32(sc + (size_t)(mt * 16 + ln) * NKP + ks * 32 + kbA);
        acc[mt] = wmma_bf16(av, bf.v, acc[mt]);
      }
    }
#pragma unroll
    for (int mt = 0; mt < 2; ++mt)
#pragma unroll
      for (int r = 0; r < 8; ++r) {
        const int m = mt * 16 + mhi + r;
        CTX[(size_t)(bI * TT + q0 + m) * NFC + h * DKC + wid * 16 + ln] =
            (__bf16)acc[mt][r];
      }
  }
}

// ---------------------------------------------------------------------------
extern "C" void kernel_launch(void* const* d_in, const int* in_sizes, int n_in,
                              void* d_out, int out_size, void* d_ws, size_t ws_size,
                              hipStream_t stream) {
  const float* query   = (const float*)d_in[0];
  const float* key     = (const float*)d_in[1];
  const float* value   = (const float*)d_in[2];
  const unsigned char* pad = (const unsigned char*)d_in[3];
  const float* pos_emb = (const float*)d_in[4];
  const float* Wq   = (const float*)d_in[5];
  const float* bq   = (const float*)d_in[6];
  const float* Wk   = (const float*)d_in[7];
  const float* bk   = (const float*)d_in[8];
  const float* Wv   = (const float*)d_in[9];
  const float* bv   = (const float*)d_in[10];
  const float* Wpos = (const float*)d_in[11];
  const float* pbu  = (const float*)d_in[12];
  const float* pbv  = (const float*)d_in[13];
  const float* Wo   = (const float*)d_in[14];
  const float* bo   = (const float*)d_in[15];

  const size_t SZ  = (size_t)BB * TT * NFC;  // 4,194,304
  const size_t WSZ = (size_t)NFC * NFC;      // 262,144
  const size_t PE  = (size_t)BAND * NFC;     // 131,584

  // workspace layout: all bf16 except bias vectors. ~69 MB.
  __bf16* qb  = (__bf16*)d_ws;
  __bf16* kb  = qb  + SZ;
  __bf16* vb  = kb  + SZ;
  __bf16* peb = vb  + SZ;
  __bf16* wqb = peb + PE;
  __bf16* wkb = wqb + WSZ;
  __bf16* wvb = wkb + WSZ;
  __bf16* wpb = wvb + WSZ;
  __bf16* wob = wpb + WSZ;
  __bf16* QU  = wob + WSZ;
  __bf16* QV  = QU  + SZ;
  __bf16* KW  = QV  + SZ;
  __bf16* VW  = KW  + SZ;
  __bf16* PW  = VW  + SZ;                    // 272*512 reserved
  __bf16* CTX = PW  + (size_t)272 * NFC;
  float*  BU  = (float*)(CTX + SZ);
  float*  BV  = BU + NFC;

  auto cvt = [&](const float* s, __bf16* d, size_t n) {
    const int blocks = (int)((n / 8 + 255) / 256);
    cvt_f32_to_bf16<<<blocks, 256, 0, stream>>>(s, d, (int)n);
  };
  cvt(query,   qb,  SZ);
  cvt(key,     kb,  SZ);
  cvt(value,   vb,  SZ);
  cvt(pos_emb, peb, PE);
  cvt(Wq,   wqb, WSZ);
  cvt(Wk,   wkb, WSZ);
  cvt(Wv,   wvb, WSZ);
  cvt(Wpos, wpb, WSZ);
  cvt(Wo,   wob, WSZ);

  bias_combine<<<1, NFC, 0, stream>>>(bq, pbu, pbv, BU, BV);

  dim3 blk(256);
  dim3 gBig(NFC / 128, (BB * TT) / 128);        // (4, 64)
  gemm_bf16<__bf16><<<gBig, blk, 0, stream>>>(qb, NFC, wqb, NFC, BU, QU, NFC, BB * TT, NFC, NFC);
  gemm_bf16<__bf16><<<gBig, blk, 0, stream>>>(qb, NFC, wqb, NFC, BV, QV, NFC, BB * TT, NFC, NFC);
  gemm_bf16<__bf16><<<gBig, blk, 0, stream>>>(kb, NFC, wkb, NFC, bk, KW, NFC, BB * TT, NFC, NFC);
  gemm_bf16<__bf16><<<gBig, blk, 0, stream>>>(vb, NFC, wvb, NFC, bv, VW, NFC, BB * TT, NFC, NFC);
  dim3 gPos(NFC / 128, (BAND + 127) / 128);     // (4, 3)
  gemm_bf16<__bf16><<<gPos, blk, 0, stream>>>(peb, NFC, wpb, NFC, nullptr, PW, NFC, BAND, NFC, NFC);

  constexpr size_t ATTN_SMEM = (size_t)QBC * NKP * 4 + (size_t)NKC * DKC * 2 + 256 * 4; // 83968 B
  (void)hipFuncSetAttribute((const void*)attn_kernel,
                            hipFuncAttributeMaxDynamicSharedMemorySize, (int)ATTN_SMEM);
  attn_kernel<<<dim3(TT / QBC, NHEAD, BB), dim3(128), ATTN_SMEM, stream>>>(
      QU, QV, KW, VW, PW, pad, CTX);

  gemm_bf16<float><<<gBig, blk, 0, stream>>>(CTX, NFC, wob, NFC, bo, (float*)d_out,
                                             NFC, BB * TT, NFC, NFC);
}
